// LongAttention_82703890251934
// MI455X (gfx1250) — matrix operand order repaired and checked by reading, not compiled, not run on hardware
//
#include <hip/hip_runtime.h>
#include <hip/hip_bf16.h>

// ---------------------------------------------------------------------------
// Problem constants (from reference)
// ---------------------------------------------------------------------------
#define BB 2
#define SS 4096
#define DD 1024
#define HH 16
#define DHH 64
#define WW 256
#define GG 64
#define NB (SS / WW)   // 16

typedef __attribute__((ext_vector_type(16))) _Float16 v16h;
typedef __attribute__((ext_vector_type(8)))  float    v8f;
typedef __attribute__((ext_vector_type(4)))  unsigned int u32x4;

union TileA_u { v16h h; unsigned int u[8]; };
union TileB_u { v16h h; u32x4 q[2]; };

// ---------------------------------------------------------------------------
// A-operand loader: 16x32 f16 tile, source row-major (row stride ld halves).
// Lane = M row; VGPR v half p -> K = 16*hi + 2v + p:
//   lanes 0-15 : K = 0..15;  lanes 16-31 : K = 16..31
// ---------------------------------------------------------------------------
__device__ __forceinline__ v16h load_tile_A(const _Float16* p, int ld) {
  const int lane = threadIdx.x & 31;
  const int hi = lane >> 4;
  const _Float16* base = p + (size_t)(lane & 15) * ld;
  TileA_u t;
#pragma unroll
  for (int v = 0; v < 8; ++v) {
    const int kof = 2 * v + 16 * hi;
    t.u[v] = *(const unsigned int*)(base + kof);
  }
  return t.h;
}

// ---------------------------------------------------------------------------
// B-operand loader: 32x16 f16 tile in N x K form (row = N, col = K contiguous,
// row stride ld halves). Lanes 0-15: K=0..15, lanes 16-31: K=16..31;
// VGPR v half p -> K = 16*hi + 2v + p. Two 16-byte loads per lane.
// ---------------------------------------------------------------------------
__device__ __forceinline__ v16h load_tile_B(const _Float16* p, int ld) {
  const int lane = threadIdx.x & 31;
  const int hi = lane >> 4;
  const _Float16* base = p + (size_t)(lane & 15) * ld + 16 * hi;
  TileB_u t;
  t.q[0] = *(const u32x4*)(base);
  t.q[1] = *(const u32x4*)(base + 8);
  return t.h;
}

__device__ __forceinline__ v8f wmma_f16(v16h a, v16h b, v8f c) {
  return __builtin_amdgcn_wmma_f32_16x16x32_f16(
      /*neg_a=*/false, a, /*neg_b=*/false, b,
      /*c_mod=*/(short)0, c, /*reuse_a=*/false, /*reuse_b=*/false);
}

// ---------------------------------------------------------------------------
// Elementwise converts
// ---------------------------------------------------------------------------
__global__ __launch_bounds__(256) void f32_to_f16_kernel(
    const float* __restrict__ in, _Float16* __restrict__ out, int n) {
  int i = (blockIdx.x * 256 + threadIdx.x) * 4;
  if (i + 3 < n) {
#pragma unroll
    for (int j = 0; j < 4; ++j) out[i + j] = (_Float16)in[i + j];
  }
}

// WT[n*D + k] = (f16) W[k*D + n]
__global__ __launch_bounds__(256) void transpose_w_kernel(
    const float* __restrict__ Wm, _Float16* __restrict__ WT) {
  int i = blockIdx.x * 256 + threadIdx.x;          // 0 .. D*D-1
  int n = i >> 10;
  int k = i & (DD - 1);
  WT[i] = (_Float16)Wm[(size_t)k * DD + n];
}

// ---------------------------------------------------------------------------
// GEMM: C[row, n] = sum_k A[row,k] * BT[n,k]   (rows = B*S, n,k in [0,D))
// flag 0: f16 out, layout [B,H,S,DH]      (Q, K)
// flag 1: f16 out, layout [B,H,DH,S]      (V transposed)
// flag 2: f32 out, layout [B,S,D]         (final @ Wo)
// grid (B*S/32, D/256), block 256 = 8 waves arranged 2(row) x 4(col);
// each wave computes a 16x64 strip = 4 WMMA tiles, A double-buffered in LDS.
// ---------------------------------------------------------------------------
__global__ __launch_bounds__(256) void gemm_kernel(
    const _Float16* __restrict__ Ain, const _Float16* __restrict__ BT,
    _Float16* __restrict__ outH, float* __restrict__ outF, int flag) {
  __shared__ __align__(16) _Float16 As[2][32 * 32];
  const int row0 = blockIdx.x * 32;
  const int wave = threadIdx.x >> 5;
  const int wr = wave & 1;                  // row half within block
  const int wc = wave >> 1;                 // col quarter within block
  const int n0 = blockIdx.y * 256 + wc * 64;

  v8f acc[4];
#pragma unroll
  for (int j = 0; j < 4; ++j)
#pragma unroll
    for (int r = 0; r < 8; ++r) acc[j][r] = 0.f;

  const int tid = threadIdx.x;
  const int sRow = tid >> 4;                // 0..15
  const int sCol = (tid & 15) << 1;         // 0..30 even

  // prologue: stage k0 = 0 into buffer 0 (each thread: 2 dwords)
  *(unsigned int*)&As[0][sRow * 32 + sCol] =
      *(const unsigned int*)&Ain[(size_t)(row0 + sRow) * DD + sCol];
  *(unsigned int*)&As[0][(sRow + 16) * 32 + sCol] =
      *(const unsigned int*)&Ain[(size_t)(row0 + sRow + 16) * DD + sCol];
  __syncthreads();

  for (int i = 0; i < DD / 32; ++i) {
    const int k0 = i * 32;
    const int cur = i & 1;
    if (i + 1 < DD / 32) {                  // stage next tile into other buffer
      const int kn = k0 + 32;
      *(unsigned int*)&As[cur ^ 1][sRow * 32 + sCol] =
          *(const unsigned int*)&Ain[(size_t)(row0 + sRow) * DD + kn + sCol];
      *(unsigned int*)&As[cur ^ 1][(sRow + 16) * 32 + sCol] =
          *(const unsigned int*)&Ain[(size_t)(row0 + sRow + 16) * DD + kn + sCol];
      __builtin_prefetch(BT + (size_t)n0 * DD + kn, 0, 1);
    }
    v16h a  = load_tile_A(&As[cur][wr * 16 * 32], 32);
    v16h b0 = load_tile_B(BT + (size_t)(n0 + 0)  * DD + k0, DD);
    v16h b1 = load_tile_B(BT + (size_t)(n0 + 16) * DD + k0, DD);
    v16h b2 = load_tile_B(BT + (size_t)(n0 + 32) * DD + k0, DD);
    v16h b3 = load_tile_B(BT + (size_t)(n0 + 48) * DD + k0, DD);
    acc[0] = wmma_f16(a, b0, acc[0]);
    acc[1] = wmma_f16(a, b1, acc[1]);
    acc[2] = wmma_f16(a, b2, acc[2]);
    acc[3] = wmma_f16(a, b3, acc[3]);
    __syncthreads();
  }

  const int lane = tid & 31, nCol = lane & 15, hiL = lane >> 4;
#pragma unroll
  for (int j = 0; j < 4; ++j) {
#pragma unroll
    for (int r = 0; r < 8; ++r) {
      const int m = r + 8 * hiL;
      const int rowAbs = row0 + wr * 16 + m;
      const int n = n0 + j * 16 + nCol;
      if (flag == 2) {
        outF[(size_t)rowAbs * DD + n] = acc[j][r];
      } else {
        const int b = rowAbs >> 12, s = rowAbs & (SS - 1);
        const int h = n >> 6, dh = n & (DHH - 1);
        size_t idxO;
        if (flag == 0)
          idxO = ((((size_t)b * HH + h) * SS) + s) * DHH + dh;
        else
          idxO = ((((size_t)b * HH + h) * DHH) + dh) * SS + s;
        outH[idxO] = (_Float16)acc[j][r];
      }
    }
  }
}

// ---------------------------------------------------------------------------
// Attention: one block per (b, h, q-block). Flash-style online softmax over
// 32-key chunks: local band chunks (masked) then the 2 global chunks.
// ---------------------------------------------------------------------------
template <bool MASKED>
__device__ __forceinline__ void process_chunk(
    int kb, int q0, const v16h& qa0, const v16h& qa1,
    const _Float16* Kbh, const _Float16* Vbh, _Float16* Pw,
    v8f& o0, v8f& o1, v8f& o2, v8f& o3, float* mrow, float* lrow) {
  const int lane = threadIdx.x & 31;
  const int nCol = lane & 15, hiL = lane >> 4;

  v8f s0, s1;
#pragma unroll
  for (int r = 0; r < 8; ++r) { s0[r] = 0.f; s1[r] = 0.f; }

  // scores: S = Q (16x64) . K^T  -> two 16x16 tiles (keys kb..+15, kb+16..+31)
  {
    v16h kb0 = load_tile_B(Kbh + (size_t)kb * DHH + 0, DHH);
    v16h kb1 = load_tile_B(Kbh + (size_t)kb * DHH + 32, DHH);
    s0 = wmma_f16(qa0, kb0, s0);
    s0 = wmma_f16(qa1, kb1, s0);
    v16h kc0 = load_tile_B(Kbh + (size_t)(kb + 16) * DHH + 0, DHH);
    v16h kc1 = load_tile_B(Kbh + (size_t)(kb + 16) * DHH + 32, DHH);
    s1 = wmma_f16(qa0, kc0, s1);
    s1 = wmma_f16(qa1, kc1, s1);
  }

#pragma unroll
  for (int r = 0; r < 8; ++r) {
    const int m = r + 8 * hiL;
    const int qi = q0 + m;
    float a = s0[r] * 0.125f;   // 1/sqrt(DH)
    float c = s1[r] * 0.125f;
    if (MASKED) {
      const int kj0 = kb + nCol;
      const int kj1 = kb + 16 + nCol;
      int d0 = kj0 - qi; if (d0 < 0) d0 = -d0;
      int d1 = kj1 - qi; if (d1 < 0) d1 = -d1;
      if (!(kj0 >= GG && kj0 < SS && d0 <= WW)) a = -1e9f;
      if (!(kj1 >= GG && kj1 < SS && d1 <= WW)) c = -1e9f;
    }
    // per-row max across the 16 lanes holding this row's columns
    float cm = fmaxf(a, c);
    cm = fmaxf(cm, __shfl_xor(cm, 1, 32));
    cm = fmaxf(cm, __shfl_xor(cm, 2, 32));
    cm = fmaxf(cm, __shfl_xor(cm, 4, 32));
    cm = fmaxf(cm, __shfl_xor(cm, 8, 32));
    const float mNew = fmaxf(mrow[r], cm);
    const float alpha = __expf(mrow[r] - mNew);
    mrow[r] = mNew;
    const float p0 = __expf(a - mNew);
    const float p1 = __expf(c - mNew);
    float ps = p0 + p1;
    ps += __shfl_xor(ps, 1, 32);
    ps += __shfl_xor(ps, 2, 32);
    ps += __shfl_xor(ps, 4, 32);
    ps += __shfl_xor(ps, 8, 32);
    lrow[r] = lrow[r] * alpha + ps;
    o0[r] *= alpha; o1[r] *= alpha; o2[r] *= alpha; o3[r] *= alpha;
    // stage P (C-layout -> LDS row-major 16x32)
    Pw[m * 32 + nCol] = (_Float16)p0;
    Pw[m * 32 + 16 + nCol] = (_Float16)p1;
  }

  // O += P (16x32) . V (32 keys x 64 feats); B-operand from V^T rows
  v16h pa = load_tile_A(Pw, 32);
  v16h vb0 = load_tile_B(Vbh + (size_t)0  * SS + kb, SS);
  v16h vb1 = load_tile_B(Vbh + (size_t)16 * SS + kb, SS);
  v16h vb2 = load_tile_B(Vbh + (size_t)32 * SS + kb, SS);
  v16h vb3 = load_tile_B(Vbh + (size_t)48 * SS + kb, SS);
  o0 = wmma_f16(pa, vb0, o0);
  o1 = wmma_f16(pa, vb1, o1);
  o2 = wmma_f16(pa, vb2, o2);
  o3 = wmma_f16(pa, vb3, o3);
}

__global__ __launch_bounds__(256) void attn_kernel(
    const _Float16* __restrict__ Q, const _Float16* __restrict__ K,
    const _Float16* __restrict__ VT, _Float16* __restrict__ ctx) {
  __shared__ __align__(16) _Float16 Pl[8][16 * 32];  // per-wave P staging
  const int idx = blockIdx.x;    // B*H*NB blocks
  const int qb = idx & (NB - 1);
  const int bh = idx >> 4;       // b*H + h
  const int bI = bh >> 4;
  const int hI = bh & (HH - 1);

  const _Float16* Qbh = Q + (size_t)bh * SS * DHH;
  const _Float16* Kbh = K + (size_t)bh * SS * DHH;
  const _Float16* Vbh = VT + (size_t)bh * DHH * SS;

  const int wave = threadIdx.x >> 5;
  const int lane = threadIdx.x & 31;
  const int nCol = lane & 15, hiL = lane >> 4;
  _Float16* Pw = &Pl[wave][0];

  for (int t = wave * 2; t < wave * 2 + 2; ++t) {
    const int q0 = qb * WW + t * 16;  // absolute sequence row of this q-tile
    const v16h qa0 = load_tile_A(Qbh + (size_t)q0 * DHH + 0, DHH);
    const v16h qa1 = load_tile_A(Qbh + (size_t)q0 * DHH + 32, DHH);

    v8f o0, o1, o2, o3;
    float mrow[8], lrow[8];
#pragma unroll
    for (int r = 0; r < 8; ++r) {
      o0[r] = 0.f; o1[r] = 0.f; o2[r] = 0.f; o3[r] = 0.f;
      mrow[r] = -1e9f; lrow[r] = 0.f;
    }

    // local band: keys [max(qb*W - W, G), min(qb*W + 2W, S)), 32 at a time
    const int kbLo = (qb * WW - WW > GG) ? (qb * WW - WW) : GG;
    const int kbHi = (qb * WW + 2 * WW < SS) ? (qb * WW + 2 * WW) : SS;
    for (int kb = kbLo; kb < kbHi; kb += 32)
      process_chunk<true>(kb, q0, qa0, qa1, Kbh, Vbh, Pw,
                          o0, o1, o2, o3, mrow, lrow);
    // global keys [0, G): unmasked
    for (int kb = 0; kb < GG; kb += 32)
      process_chunk<false>(kb, q0, qa0, qa1, Kbh, Vbh, Pw,
                           o0, o1, o2, o3, mrow, lrow);

    // normalize + write context [B,S,D] f16
#pragma unroll
    for (int r = 0; r < 8; ++r) {
      const int m = r + 8 * hiL;
      const float inv = 1.0f / lrow[r];
      const size_t base = (((size_t)bI * SS) + q0 + m) * DD + (size_t)hI * DHH;
      ctx[base + 0  + nCol] = (_Float16)(o0[r] * inv);
      ctx[base + 16 + nCol] = (_Float16)(o1[r] * inv);
      ctx[base + 32 + nCol] = (_Float16)(o2[r] * inv);
      ctx[base + 48 + nCol] = (_Float16)(o3[r] * inv);
    }
  }
}

// ---------------------------------------------------------------------------
// Host launcher
// ---------------------------------------------------------------------------
extern "C" void kernel_launch(void* const* d_in, const int* in_sizes, int n_in,
                              void* d_out, int out_size, void* d_ws, size_t ws_size,
                              hipStream_t stream) {
  (void)in_sizes; (void)n_in; (void)out_size; (void)ws_size;
  const float* x  = (const float*)d_in[0];
  const float* Wq = (const float*)d_in[1];
  const float* Wk = (const float*)d_in[2];
  const float* Wv = (const float*)d_in[3];
  const float* Wo = (const float*)d_in[4];
  float* out = (float*)d_out;

  const size_t NX = (size_t)BB * SS * DD;   // 8388608
  const size_t NW = (size_t)DD * DD;        // 1048576

  char* ws = (char*)d_ws;
  size_t off = 0;
  auto take = [&](size_t halves) -> _Float16* {
    _Float16* p = (_Float16*)(ws + off);
    off += halves * sizeof(_Float16);
    return p;
  };
  _Float16* xh   = take(NX);
  _Float16* wqT  = take(NW);
  _Float16* wkT  = take(NW);
  _Float16* wvT  = take(NW);
  _Float16* woT  = take(NW);
  _Float16* Qh   = take(NX);
  _Float16* Kh   = take(NX);
  _Float16* VTh  = take(NX);
  _Float16* ctxh = take(NX);

  // 1) convert inputs
  f32_to_f16_kernel<<<(unsigned)(NX / (256 * 4)), 256, 0, stream>>>(x, xh, (int)NX);
  transpose_w_kernel<<<(unsigned)(NW / 256), 256, 0, stream>>>(Wq, wqT);
  transpose_w_kernel<<<(unsigned)(NW / 256), 256, 0, stream>>>(Wk, wkT);
  transpose_w_kernel<<<(unsigned)(NW / 256), 256, 0, stream>>>(Wv, wvT);
  transpose_w_kernel<<<(unsigned)(NW / 256), 256, 0, stream>>>(Wo, woT);

  // 2) projections: grid (B*S/32, D/256)
  dim3 gg((unsigned)(BB * SS / 32), (unsigned)(DD / 256));
  gemm_kernel<<<gg, 256, 0, stream>>>(xh, wqT, Qh,  nullptr, 0);
  gemm_kernel<<<gg, 256, 0, stream>>>(xh, wkT, Kh,  nullptr, 0);
  gemm_kernel<<<gg, 256, 0, stream>>>(xh, wvT, VTh, nullptr, 1);

  // 3) banded + global attention
  attn_kernel<<<(unsigned)(BB * HH * NB), 256, 0, stream>>>(Qh, Kh, VTh, ctxh);

  // 4) output projection (fp32 out)
  gemm_kernel<<<gg, 256, 0, stream>>>(ctxh, woT, nullptr, out, 2);
}
